// GNNCell_51539608062
// MI455X (gfx1250) — compile-verified, matrix operand order checked by reading, not run on hardware
//
#include <hip/hip_runtime.h>

typedef __bf16 bf16_t;
typedef bf16_t v16bf __attribute__((ext_vector_type(16)));
typedef bf16_t v8bf  __attribute__((ext_vector_type(8)));
typedef float  v8f   __attribute__((ext_vector_type(8)));
typedef unsigned v4u __attribute__((ext_vector_type(4)));
typedef int      v4i __attribute__((ext_vector_type(4)));
typedef int      v8i __attribute__((ext_vector_type(8)));

constexpr int KN = 100000;   // nodes
constexpr int KD = 16;       // neighbors per node
constexpr int KH = 128;      // hidden
constexpr int KC = 10;       // class outputs
constexpr int TM = 16;       // node tile per workgroup
constexpr int G4H = 4 * KH;  // 512 gate columns

// Tensor Data Mover builtin arity differs between toolchains (probe-verified):
// ROCm 7.2 / clang-22: 5 args; amdgpu-toolchain / clang-23: 6 args.
#if defined(__clang_major__) && __clang_major__ >= 23
#define TDM_CALL(g0, g1, g2, g3) \
  __builtin_amdgcn_tensor_load_to_lds((g0), (g1), (g2), (g3), (v8i){0, 0, 0, 0, 0, 0, 0, 0}, 0)
#else
#define TDM_CALL(g0, g1, g2, g3) \
  __builtin_amdgcn_tensor_load_to_lds((g0), (g1), (g2), (g3), 0)
#endif

__device__ __forceinline__ float sigm(float x) { return 1.0f / (1.0f + __expf(-x)); }
__device__ __forceinline__ float tanh_f(float x) { return 2.0f / (1.0f + __expf(-2.0f * x)) - 1.0f; }

// Load one 16x32 bf16 WMMA operand fragment from row-major [rows][128] data.
// ISA layout: lanes 0-15 row M=lane, K=0..7 (v0-3) & K=16..23 (v4-7);
// lanes 16-31 same row, K=8..15 & K=24..31.  kc selects the 32-wide K chunk.
__device__ __forceinline__ v16bf ldfrag(const bf16_t* base, int lane, int kc) {
  const bf16_t* p = base + ((lane & 15) << 7) + kc * 32 + ((lane >> 4) << 3);
  v8bf lo = *(const v8bf*)p;          // ds/global load b128
  v8bf hi = *(const v8bf*)(p + 16);   // ds/global load b128
  return __builtin_shufflevector(lo, hi, 0, 1, 2, 3, 4, 5, 6, 7, 8, 9, 10, 11, 12, 13, 14, 15);
}

__device__ __forceinline__ v8f wmma_bf16(v16bf a, v16bf b, v8f c) {
  return __builtin_amdgcn_wmma_f32_16x16x32_bf16(false, a, false, b, (short)0, c, false, false);
}

// TDM gather-mode load: 8 rows of h (bf16, 128 wide = 256B each) selected by
// 32-bit row indices, DMA'd from global/L2 into LDS at lds_addr. D# layout per
// CDNA5 ISA (gather_mode=1, gather_index_size=1(32b),
// data_size=1(2B), tile_dim0=128, tile_dim1=8 valid indices, dim0_stride=128).
__device__ __forceinline__ void tdm_gather8(unsigned lds_addr, unsigned long long gaddr,
                                            const int* ids, v8i g1) {
  v4u g0 = {0xC0000001u,  // gather_mode | 32-bit indices | count=1
            lds_addr,
            (unsigned)(gaddr & 0xFFFFFFFFull),
            (unsigned)(((gaddr >> 32) & 0x01FFFFFFull) | (2u << 30))};  // type=2 ("image")
  v4i g2 = {ids[0], ids[1], ids[2], ids[3]};
  v4i g3 = {ids[4], ids[5], ids[6], ids[7]};
  TDM_CALL(g0, g1, g2, g3);
}

__global__ __launch_bounds__(256) void cvt_f32_to_bf16(const float* __restrict__ s,
                                                       bf16_t* __restrict__ d, int n) {
  int i = blockIdx.x * 256 + threadIdx.x;
  if (i < n) d[i] = (bf16_t)s[i];
}

// One SAGEConv(lstm) layer over a 16-node tile per workgroup.
// LDS: xbuf0 | xbuf1 (double-buffered TDM gather target) | hsbuf | selfbuf.
__global__ __launch_bounds__(256) void sage_lstm_layer(
    const bf16_t* __restrict__ hin,     // [N][128] bf16
    const int*    __restrict__ nbr,     // [N][16]
    const bf16_t* __restrict__ wih,     // [512][128] bf16
    const bf16_t* __restrict__ whh,     // [512][128] bf16
    const float*  __restrict__ bvec,    // [512]
    const bf16_t* __restrict__ wself,   // [128][128] bf16
    const float*  __restrict__ bself,   // [128]
    const bf16_t* __restrict__ wneigh,  // [128][128] bf16
    const float*  __restrict__ bneigh,  // [128]
    bf16_t*       __restrict__ hout)    // [N][128] bf16
{
  extern __shared__ char smem[];
  bf16_t* xbuf0   = (bf16_t*)smem;          // [16][128] gather buffer, parity 0
  bf16_t* xbuf1   = xbuf0 + TM * KH;        // [16][128] gather buffer, parity 1
  bf16_t* hsbuf   = xbuf1 + TM * KH;        // [16][128] LSTM hidden state
  bf16_t* selfbuf = hsbuf + TM * KH;        // [16][128] tile's own features

  const int tid  = threadIdx.x;
  const int lane = tid & 31;
  const int w    = tid >> 5;                // wave id 0..7 -> hidden slice [16w,16w+16)
  const int col  = lane & 15;
  const int half = lane >> 4;
  const int node0 = blockIdx.x * TM;

  // LDS byte address of the gather buffers (flat LDS addr = low 32 bits).
  const unsigned xoff0 = (unsigned)(unsigned long long)(void*)xbuf0;
  const unsigned long long hbase = (unsigned long long)hin;

  // Stage the tile's own rows; zero the LSTM hidden state buffer.
  {
    int m = tid >> 4, ch = tid & 15;
    *(uint4*)(selfbuf + m * KH + ch * 8) =
        *(const uint4*)(hin + (size_t)(node0 + m) * KH + ch * 8);
    *(uint4*)(hsbuf + m * KH + ch * 8) = make_uint4(0u, 0u, 0u, 0u);
  }

  // Constant part of the TDM descriptor (group 1).
  v8i g1 = {0, 0, 0, 0, 0, 0, 0, 0};
  g1[0] = 1 << 16;                                   // data_size = 2 bytes
  g1[1] = (int)((KH & 0xFFFF) << 16);                // tensor_dim0[63:48] = 128
  g1[2] = (int)((unsigned)(KN & 0xFFFF) << 16);      // tensor_dim1[95:80]
  g1[3] = (int)((KN >> 16) | (KH << 16));            // tensor_dim1[111:96] | tile_dim0=128
  g1[4] = 8;                                         // tile_dim1 = 8 valid indices
  g1[5] = KH;                                        // tensor_dim0_stride = 128

  // Prologue: wave 0 kicks off the TDM gather for step 0 (parity 0).
  if (w == 0) {
    int ids[16];
#pragma unroll
    for (int i = 0; i < 16; ++i) ids[i] = nbr[(size_t)(node0 + i) * KD];  // t = 0
    tdm_gather8(xoff0, hbase, ids + 0, g1);
    tdm_gather8(xoff0 + 2048, hbase, ids + 8, g1);
  }
  __syncthreads();  // hsbuf zeros + selfbuf visible

  // Per-lane gate biases for this wave's slice.
  float bi[4];
#pragma unroll
  for (int g = 0; g < 4; ++g) bi[g] = bvec[g * KH + w * 16 + col];

  // Recurrent weights: all Whh B-fragments for this wave in VGPRs (reused 16x).
  v16bf whhf[4][4];
#pragma unroll
  for (int g = 0; g < 4; ++g)
#pragma unroll
    for (int kc = 0; kc < 4; ++kc)
      whhf[g][kc] = ldfrag(whh + (size_t)(g * KH + w * 16) * KH, lane, kc);

  v8f cs = {0.f, 0.f, 0.f, 0.f, 0.f, 0.f, 0.f, 0.f};

  for (int t = 0; t < KD; ++t) {
    // Wave 0: make sure the TDM gather for step t has landed in LDS.
    if (w == 0) __builtin_amdgcn_s_wait_tensorcnt(0);
    __syncthreads();  // publish xbuf[t&1]; also previous step's hsbuf writes

    // Wave 0: launch the gather for step t+1 into the other parity buffer
    // (overlaps with this step's WMMA work on all waves).
    if (w == 0 && t + 1 < KD) {
      int ids[16];
#pragma unroll
      for (int i = 0; i < 16; ++i) ids[i] = nbr[(size_t)(node0 + i) * KD + (t + 1)];
      unsigned poff = xoff0 + (((t + 1) & 1) ? (unsigned)(TM * KH * 2) : 0u);
      tdm_gather8(poff, hbase, ids + 0, g1);
      tdm_gather8(poff + 2048, hbase, ids + 8, g1);
    }

    const bf16_t* xcur = (t & 1) ? xbuf1 : xbuf0;
    v16bf xa[4], ha[4];
#pragma unroll
    for (int kc = 0; kc < 4; ++kc) {
      xa[kc] = ldfrag(xcur, lane, kc);
      ha[kc] = ldfrag(hsbuf, lane, kc);
    }

    v8f acc[4];
#pragma unroll
    for (int g = 0; g < 4; ++g) {
      v8f a = {0.f, 0.f, 0.f, 0.f, 0.f, 0.f, 0.f, 0.f};
#pragma unroll
      for (int kc = 0; kc < 4; ++kc) {
        v16bf bw = ldfrag(wih + (size_t)(g * KH + w * 16) * KH, lane, kc);  // re-read from global every step
        a = wmma_bf16(xa[kc], bw, a);
        a = wmma_bf16(ha[kc], whhf[g][kc], a);
      }
      acc[g] = a;
    }
    __syncthreads();  // all reads of xbuf/hsbuf done before hsbuf overwrite

    // Elementwise LSTM cell update; wave-local: lane owns column j = 16w+col.
#pragma unroll
    for (int r = 0; r < 8; ++r) {
      float iv = sigm(acc[0][r] + bi[0]);
      float fv = sigm(acc[1][r] + bi[1]);
      float gv = tanh_f(acc[2][r] + bi[2]);
      float ov = sigm(acc[3][r] + bi[3]);
      float c  = fv * cs[r] + iv * gv;
      cs[r] = c;
      float hv = ov * tanh_f(c);
      hsbuf[(half * 8 + r) * KH + w * 16 + col] = (bf16_t)hv;
    }
  }
  __syncthreads();

  // h_out = relu(h_self @ Wself^T + bself + h_neigh @ Wneigh^T + bneigh)
  v16bf sa[4], na[4];
#pragma unroll
  for (int kc = 0; kc < 4; ++kc) {
    sa[kc] = ldfrag(selfbuf, lane, kc);
    na[kc] = ldfrag(hsbuf, lane, kc);
  }
  v8f acc = {0.f, 0.f, 0.f, 0.f, 0.f, 0.f, 0.f, 0.f};
#pragma unroll
  for (int kc = 0; kc < 4; ++kc) {
    acc = wmma_bf16(sa[kc], ldfrag(wself + (size_t)(w * 16) * KH, lane, kc), acc);
    acc = wmma_bf16(na[kc], ldfrag(wneigh + (size_t)(w * 16) * KH, lane, kc), acc);
  }
  float bb = bself[w * 16 + col] + bneigh[w * 16 + col];
#pragma unroll
  for (int r = 0; r < 8; ++r) {
    float v = acc[r] + bb;
    v = v > 0.f ? v : 0.f;
    hout[(size_t)(node0 + half * 8 + r) * KH + w * 16 + col] = (bf16_t)v;
  }
}

// One hidden MLP stage: dst = relu(src @ W[li]^T + b[li]), 16x128 tile, WMMA.
__device__ __forceinline__ void mlp_stage(const bf16_t* __restrict__ Wb,
                                          const float* __restrict__ bv,
                                          const bf16_t* src, bf16_t* dst,
                                          int lane, int w, int col, int half, int li) {
  v16bf xa[4];
#pragma unroll
  for (int kc = 0; kc < 4; ++kc) xa[kc] = ldfrag(src, lane, kc);
  v8f acc = {0.f, 0.f, 0.f, 0.f, 0.f, 0.f, 0.f, 0.f};
#pragma unroll
  for (int kc = 0; kc < 4; ++kc) {
    v16bf bw = ldfrag(Wb + (size_t)li * KH * KH + (size_t)(w * 16) * KH, lane, kc);
    acc = wmma_bf16(xa[kc], bw, acc);
  }
  float bb = bv[li * KH + w * 16 + col];
#pragma unroll
  for (int r = 0; r < 8; ++r) {
    float v = acc[r] + bb;
    v = v > 0.f ? v : 0.f;
    dst[(half * 8 + r) * KH + w * 16 + col] = (bf16_t)v;
  }
}

__global__ __launch_bounds__(256) void heads_kernel(
    const bf16_t* __restrict__ hfin,                        // final h, bf16
    const bf16_t* __restrict__ clsWb, const float* __restrict__ clsb,
    const float*  __restrict__ clsoW, const float* __restrict__ clsob,
    const bf16_t* __restrict__ cnfWb, const float* __restrict__ cnfb,
    const float*  __restrict__ cnfoW, const float* __restrict__ cnfob,
    float* __restrict__ out)                                // [o | h | l]
{
  extern __shared__ char smem[];
  bf16_t* hbuf = (bf16_t*)smem;       // [16][128]
  bf16_t* xb0  = hbuf + TM * KH;      // ping
  bf16_t* xb1  = xb0 + TM * KH;       // pong

  const int tid  = threadIdx.x;
  const int lane = tid & 31;
  const int w    = tid >> 5;
  const int col  = lane & 15;
  const int half = lane >> 4;
  const int node0 = blockIdx.x * TM;

  const size_t o_off = 0;
  const size_t h_off = (size_t)KN * KC;
  const size_t l_off = h_off + (size_t)KN * KH;

  // Stage final h into LDS and emit the fp32 h output.
  {
    int m = tid >> 4, ch = tid & 15;
    uint4 raw = *(const uint4*)(hfin + (size_t)(node0 + m) * KH + ch * 8);
    *(uint4*)(hbuf + m * KH + ch * 8) = raw;
    const bf16_t* q = (const bf16_t*)&raw;
    float* ho = out + h_off + (size_t)(node0 + m) * KH + ch * 8;
#pragma unroll
    for (int i = 0; i < 8; ++i) ho[i] = (float)q[i];
  }

  // ---- cls head: 5x (Linear+ReLU) via WMMA, then H->C dots ----
  {
    const bf16_t* src = hbuf;
    bf16_t* dst = xb0;
#pragma unroll
    for (int li = 0; li < 5; ++li) {
      __syncthreads();
      mlp_stage(clsWb, clsb, src, dst, lane, w, col, half, li);
      src = dst;
      dst = (dst == xb0) ? xb1 : xb0;
    }
    __syncthreads();
    int m = tid >> 4, c = tid & 15;
    if (c < KC) {
      float s = clsob[c];
      for (int k = 0; k < KH; ++k) s += (float)src[m * KH + k] * clsoW[c * KH + k];
      out[o_off + (size_t)(node0 + m) * KC + c] = s;
    }
  }
  __syncthreads();

  // ---- cnf head: same pipeline, final Linear H->1 ----
  {
    const bf16_t* src = hbuf;
    bf16_t* dst = xb0;
#pragma unroll
    for (int li = 0; li < 5; ++li) {
      __syncthreads();
      mlp_stage(cnfWb, cnfb, src, dst, lane, w, col, half, li);
      src = dst;
      dst = (dst == xb0) ? xb1 : xb0;
    }
    __syncthreads();
    if (tid < TM) {
      int m = tid;
      float s = cnfob[0];
      for (int k = 0; k < KH; ++k) s += (float)src[m * KH + k] * cnfoW[k];
      out[l_off + (size_t)(node0 + m)] = s;
    }
  }
}

extern "C" void kernel_launch(void* const* d_in, const int* in_sizes, int n_in,
                              void* d_out, int out_size, void* d_ws, size_t ws_size,
                              hipStream_t stream) {
  const float* h0     = (const float*)d_in[0];
  const int*   nbr    = (const int*)d_in[1];
  const float* wih    = (const float*)d_in[2];   // [2][512][128]
  const float* whh    = (const float*)d_in[3];   // [2][512][128]
  const float* lb     = (const float*)d_in[4];   // [2][512]
  const float* wself  = (const float*)d_in[5];   // [2][128][128]
  const float* bself  = (const float*)d_in[6];   // [2][128]
  const float* wneigh = (const float*)d_in[7];   // [2][128][128]
  const float* bneigh = (const float*)d_in[8];   // [2][128]
  const float* clsW   = (const float*)d_in[9];   // [5][128][128]
  const float* clsb   = (const float*)d_in[10];
  const float* clsoW  = (const float*)d_in[11];
  const float* clsob  = (const float*)d_in[12];
  const float* cnfW   = (const float*)d_in[13];
  const float* cnfb   = (const float*)d_in[14];
  const float* cnfoW  = (const float*)d_in[15];
  const float* cnfob  = (const float*)d_in[16];
  float* out = (float*)d_out;

  char* ws = (char*)d_ws;
  size_t off = 0;
  auto walloc = [&](size_t bytes) -> char* {
    char* p = ws + off;
    off += (bytes + 255) & ~(size_t)255;
    return p;
  };
  bf16_t* hA     = (bf16_t*)walloc((size_t)KN * KH * 2);
  bf16_t* hB     = (bf16_t*)walloc((size_t)KN * KH * 2);
  bf16_t* wih_b  = (bf16_t*)walloc((size_t)2 * G4H * KH * 2);
  bf16_t* whh_b  = (bf16_t*)walloc((size_t)2 * G4H * KH * 2);
  bf16_t* wsf_b  = (bf16_t*)walloc((size_t)2 * KH * KH * 2);
  bf16_t* wng_b  = (bf16_t*)walloc((size_t)2 * KH * KH * 2);
  bf16_t* clsW_b = (bf16_t*)walloc((size_t)5 * KH * KH * 2);
  bf16_t* cnfW_b = (bf16_t*)walloc((size_t)5 * KH * KH * 2);

  auto cvt = [&](const float* s, bf16_t* d, int n) {
    cvt_f32_to_bf16<<<(n + 255) / 256, 256, 0, stream>>>(s, d, n);
  };
  cvt(wih,    wih_b,  2 * G4H * KH);
  cvt(whh,    whh_b,  2 * G4H * KH);
  cvt(wself,  wsf_b,  2 * KH * KH);
  cvt(wneigh, wng_b,  2 * KH * KH);
  cvt(clsW,   clsW_b, 5 * KH * KH);
  cvt(cnfW,   cnfW_b, 5 * KH * KH);
  cvt(h0,     hA,     KN * KH);

  const int ntiles = KN / TM;                        // 6250
  const size_t lds_layer = (size_t)4 * TM * KH * 2;  // 16 KB (2x xbuf, hsbuf, selfbuf)
  const size_t lds_head  = (size_t)3 * TM * KH * 2;  // 12 KB

  // Layer 0: hA -> hB
  sage_lstm_layer<<<ntiles, 256, lds_layer, stream>>>(
      hA, nbr,
      wih_b, whh_b, lb,
      wsf_b, bself, wng_b, bneigh,
      hB);
  // Layer 1: hB -> hA
  sage_lstm_layer<<<ntiles, 256, lds_layer, stream>>>(
      hB, nbr,
      wih_b + (size_t)G4H * KH, whh_b + (size_t)G4H * KH, lb + G4H,
      wsf_b + (size_t)KH * KH, bself + KH, wng_b + (size_t)KH * KH, bneigh + KH,
      hA);

  heads_kernel<<<ntiles, 256, lds_head, stream>>>(
      hA,
      clsW_b, clsb, clsoW, clsob,
      cnfW_b, cnfb, cnfoW, cnfob,
      out);

  (void)in_sizes; (void)n_in; (void)out_size; (void)ws_size;
}
